// MinkUNet_13657996001390
// MI455X (gfx1250) — matrix operand order repaired and checked by reading, not compiled, not run on hardware
//
#include <hip/hip_runtime.h>
#include <hip/hip_bf16.h>

// ---------------------------------------------------------------------------
// MinkUNet forward on MI455X (gfx1250).
// All conv/deconv run as bf16 implicit GEMM on v_wmma_f32_16x16x32_bf16.
// Fast kernel: one wave owns a 32-voxel x 32-cout tile; taps are walked with a
// division-free tz/ty/tx loop (neighbor base computed once per tap) and the
// inner cin/32 loop is pure b128 loads + 4 WMMAs + pointer bumps.
// ---------------------------------------------------------------------------

typedef __attribute__((ext_vector_type(16))) __bf16       v16bf;
typedef __attribute__((ext_vector_type(8)))  float        v8f;
typedef __attribute__((ext_vector_type(4)))  unsigned int u32x4;

union Frag {
    v16bf  h;
    u32x4  u[2];
    __bf16 e[16];
};

__device__ inline u32x4 zero4() {
    u32x4 z;
    z[0] = 0u; z[1] = 0u; z[2] = 0u; z[3] = 0u;
    return z;
}

struct ConvP {
    const __bf16* x;    // input activations bf16 [nvox_in, cin]
    const __bf16* wpk;  // packed weights (WMMA B fragments)
    float*        y;    // output fp32 [nvox_out, cout]
    int Zi, Yi, Xi;     // input dims  (deconv: coarse dims)
    int Zo, Yo, Xo;     // output dims (deconv: fine dims)
    int cin, cout;
    int ks, stride, pad;  // conv geometry
    int cchunks;          // cin/32   (fast kernel)
    int ktot, kchunks;    // taps*cin, ceil(/32) (generic kernel)
    int ntiles;           // cout/16  (packed-weight addressing)
    int tz, ty, tx;       // deconv tap parity
};

// ---------------------------------------------------------------------------
// Fast kernel: requires cin % 32 == 0 (all layers except the 4-channel stem).
// A fragment (16x32 bf16): lane half h = lane>>4, row M = lane&15;
//   elements 0..7  <-> K = kbase + h*8       (one b128)
//   elements 8..15 <-> K = kbase + 16 + h*8  (one b128)
// B fragment: packed [kchunk][ntile][lane][16] -> two contiguous b128 loads.
// C/D f32: element r -> M = r + h*8, N = lane&15.
// ---------------------------------------------------------------------------
template <bool DECONV>
__global__ __launch_bounds__(32) void k_conv_fast(ConvP p) {
    const int lane  = threadIdx.x;
    const int half  = lane >> 4;
    const int row   = lane & 15;
    const int col   = lane & 15;
    const int nt0   = blockIdx.y * 2;
    const int vbase = blockIdx.x * 32;

    // the two A rows this lane gathers for (voxel tile 0 and 1)
    const int Xd = DECONV ? p.Xi : p.Xo;
    const int Yd = DECONV ? p.Yi : p.Yo;
    int cx[2], cy[2], cz[2];
#pragma unroll
    for (int j = 0; j < 2; ++j) {
        const int v = vbase + j * 16 + row;
        cx[j] = v % Xd;
        cy[j] = (v / Xd) % Yd;
        cz[j] = v / (Xd * Yd);
    }

    v8f acc00, acc01, acc10, acc11;
#pragma unroll
    for (int i = 0; i < 8; ++i) {
        acc00[i] = 0.f; acc01[i] = 0.f; acc10[i] = 0.f; acc11[i] = 0.f;
    }

    const int cofs = half * 8;  // this lane's K offset within a 32-chunk

    if (DECONV) {
        const __bf16* r0 =
            p.x + (size_t)((cz[0] * p.Yi + cy[0]) * p.Xi + cx[0]) * p.cin + cofs;
        const __bf16* r1 =
            p.x + (size_t)((cz[1] * p.Yi + cy[1]) * p.Xi + cx[1]) * p.cin + cofs;
        const __bf16* bb = p.wpk + ((size_t)nt0 * 32 + lane) * 16;
        for (int cc = 0; cc < p.cchunks; ++cc) {
            Frag a0, a1, b0, b1;
            a0.u[0] = *(const u32x4*)(r0);
            a0.u[1] = *(const u32x4*)(r0 + 16);
            a1.u[0] = *(const u32x4*)(r1);
            a1.u[1] = *(const u32x4*)(r1 + 16);
            b0.u[0] = *(const u32x4*)(bb);
            b0.u[1] = *(const u32x4*)(bb + 8);
            b1.u[0] = *(const u32x4*)(bb + 512);
            b1.u[1] = *(const u32x4*)(bb + 520);
            acc00 = __builtin_amdgcn_wmma_f32_16x16x32_bf16(
                false, a0.h, false, b0.h, (short)0, acc00, false, false);
            acc01 = __builtin_amdgcn_wmma_f32_16x16x32_bf16(
                false, a0.h, false, b1.h, (short)0, acc01, false, false);
            acc10 = __builtin_amdgcn_wmma_f32_16x16x32_bf16(
                false, a1.h, false, b0.h, (short)0, acc10, false, false);
            acc11 = __builtin_amdgcn_wmma_f32_16x16x32_bf16(
                false, a1.h, false, b1.h, (short)0, acc11, false, false);
            r0 += 32;
            r1 += 32;
            bb += (size_t)p.ntiles * 512;  // next kchunk
        }
    } else {
        int tap = 0;
        for (int tzk = 0; tzk < p.ks; ++tzk)
            for (int tyk = 0; tyk < p.ks; ++tyk)
                for (int txk = 0; txk < p.ks; ++txk) {
                    // neighbor row base per voxel tile (computed once per tap)
                    bool          val[2];
                    const __bf16* rb[2];
#pragma unroll
                    for (int j = 0; j < 2; ++j) {
                        const int ix = cx[j] * p.stride + txk - p.pad;
                        const int iy = cy[j] * p.stride + tyk - p.pad;
                        const int iz = cz[j] * p.stride + tzk - p.pad;
                        val[j] = (unsigned)ix < (unsigned)p.Xi &&
                                 (unsigned)iy < (unsigned)p.Yi &&
                                 (unsigned)iz < (unsigned)p.Zi;
                        rb[j] = p.x +
                                (val[j]
                                     ? (size_t)((iz * p.Yi + iy) * p.Xi + ix) * p.cin
                                     : (size_t)0) +
                                cofs;
                    }
                    const __bf16* bb =
                        p.wpk +
                        (((size_t)tap * p.cchunks * p.ntiles + nt0) * 32 + lane) * 16;
                    for (int cc = 0; cc < p.cchunks; ++cc) {
                        Frag a0, a1, b0, b1;
                        a0.u[0] = val[0] ? *(const u32x4*)(rb[0]) : zero4();
                        a0.u[1] = val[0] ? *(const u32x4*)(rb[0] + 16) : zero4();
                        a1.u[0] = val[1] ? *(const u32x4*)(rb[1]) : zero4();
                        a1.u[1] = val[1] ? *(const u32x4*)(rb[1] + 16) : zero4();
                        b0.u[0] = *(const u32x4*)(bb);
                        b0.u[1] = *(const u32x4*)(bb + 8);
                        b1.u[0] = *(const u32x4*)(bb + 512);
                        b1.u[1] = *(const u32x4*)(bb + 520);
                        acc00 = __builtin_amdgcn_wmma_f32_16x16x32_bf16(
                            false, a0.h, false, b0.h, (short)0, acc00, false, false);
                        acc01 = __builtin_amdgcn_wmma_f32_16x16x32_bf16(
                            false, a0.h, false, b1.h, (short)0, acc01, false, false);
                        acc10 = __builtin_amdgcn_wmma_f32_16x16x32_bf16(
                            false, a1.h, false, b0.h, (short)0, acc10, false, false);
                        acc11 = __builtin_amdgcn_wmma_f32_16x16x32_bf16(
                            false, a1.h, false, b1.h, (short)0, acc11, false, false);
                        rb[0] += 32;
                        rb[1] += 32;
                        bb += (size_t)p.ntiles * 512;
                    }
                    ++tap;
                }
    }

    // epilogue: scatter C tiles
#pragma unroll
    for (int r = 0; r < 8; ++r) {
        const int m = half * 8 + r;
#pragma unroll
        for (int j = 0; j < 2; ++j) {
            const int u = vbase + j * 16 + m;
            size_t    orow;
            if (!DECONV) {
                orow = (size_t)u;
            } else {
                const int ux = u % p.Xi;
                const int uy = (u / p.Xi) % p.Yi;
                const int uz = u / (p.Xi * p.Yi);
                orow = (size_t)(((uz * 2 + p.tz) * p.Yo + (uy * 2 + p.ty)) * p.Xo +
                                (ux * 2 + p.tx));
            }
            float* o = p.y + orow * p.cout + nt0 * 16 + col;
            o[0]  = j ? acc10[r] : acc00[r];
            o[16] = j ? acc11[r] : acc01[r];
        }
    }
}

// ---------------------------------------------------------------------------
// Generic gather kernel (stem only: cin=4). One 16-voxel x 32-cout tile.
// ---------------------------------------------------------------------------
__global__ __launch_bounds__(32) void k_conv_generic(ConvP p) {
    const int lane  = threadIdx.x;
    const int half  = lane >> 4;
    const int row   = lane & 15;
    const int mtile = blockIdx.x;
    const int nt0   = blockIdx.y * 2;

    const int v  = mtile * 16 + row;
    const int vx = v % p.Xo;
    const int vy = (v / p.Xo) % p.Yo;
    const int vz = v / (p.Xo * p.Yo);

    v8f acc0, acc1;
#pragma unroll
    for (int i = 0; i < 8; ++i) { acc0[i] = 0.f; acc1[i] = 0.f; }

    for (int kc = 0; kc < p.kchunks; ++kc) {
        Frag a;
#pragma unroll
        for (int rr = 0; rr < 2; ++rr) {
            const int k0 = kc * 32 + rr * 16 + half * 8;
            for (int j = 0; j < 8; ++j) {
                const int k = k0 + j;
                __bf16    t = (__bf16)0.f;
                if (k < p.ktot) {
                    const int tap = k / p.cin;
                    const int ci  = k - tap * p.cin;
                    const int txk = tap % p.ks;
                    const int tyk = (tap / p.ks) % p.ks;
                    const int tzk = tap / (p.ks * p.ks);
                    const int ix = vx * p.stride + txk - p.pad;
                    const int iy = vy * p.stride + tyk - p.pad;
                    const int iz = vz * p.stride + tzk - p.pad;
                    if ((unsigned)ix < (unsigned)p.Xi &&
                        (unsigned)iy < (unsigned)p.Yi &&
                        (unsigned)iz < (unsigned)p.Zi) {
                        t = p.x[(size_t)((iz * p.Yi + iy) * p.Xi + ix) * p.cin + ci];
                    }
                }
                a.e[rr * 8 + j] = t;
            }
        }
        const __bf16* bb =
            p.wpk + (((size_t)kc * p.ntiles + nt0) * 32 + lane) * 16;
        Frag b0, b1;
        b0.u[0] = *(const u32x4*)bb;
        b0.u[1] = *(const u32x4*)(bb + 8);
        b1.u[0] = *(const u32x4*)(bb + 512);
        b1.u[1] = *(const u32x4*)(bb + 520);
        acc0 = __builtin_amdgcn_wmma_f32_16x16x32_bf16(
            false, a.h, false, b0.h, (short)0, acc0, false, false);
        acc1 = __builtin_amdgcn_wmma_f32_16x16x32_bf16(
            false, a.h, false, b1.h, (short)0, acc1, false, false);
    }

    const int col = lane & 15;
#pragma unroll
    for (int r = 0; r < 8; ++r) {
        const int u = mtile * 16 + half * 8 + r;
        float*    o = p.y + (size_t)u * p.cout + nt0 * 16 + col;
        o[0]  = acc0[r];
        o[16] = acc1[r];
    }
}

// ---------------------------------------------------------------------------
// Weight packing: fp32 [taps*cin, cout] -> bf16 WMMA B fragments
// layout [kchunk][ntile][lane][16], lane half h: K_local = (e/8)*16 + h*8 + e%8.
// ---------------------------------------------------------------------------
__global__ void k_pack_w(const float* w, __bf16* dst, int ktot, int kchunks,
                         int cout) {
    const size_t total = (size_t)kchunks * (cout / 16) * 512;
    const size_t i = (size_t)blockIdx.x * blockDim.x + threadIdx.x;
    if (i >= total) return;
    const int e    = (int)(i & 15);
    const int lane = (int)((i >> 4) & 31);
    const int frag = (int)(i >> 9);
    const int nt   = frag % (cout / 16);
    const int kc   = frag / (cout / 16);
    const int k    = kc * 32 + (e >> 3) * 16 + (lane >> 4) * 8 + (e & 7);
    const int co   = nt * 16 + (lane & 15);
    const float vv = (k < ktot) ? w[(size_t)k * cout + co] : 0.f;
    dst[i] = (__bf16)vv;
}

// --------------------------- small helper kernels ---------------------------
__global__ void k_mask_to_float(const int* m, float* mf, int n) {
    int i = blockIdx.x * blockDim.x + threadIdx.x;
    if (i < n) mf[i] = (float)m[i];
}

__global__ void k_pool2(const float* s, float* d, int Z, int Y, int X) {
    const int Zo = Z / 2, Yo = Y / 2, Xo = X / 2;
    const int n = Zo * Yo * Xo;
    int i = blockIdx.x * blockDim.x + threadIdx.x;
    if (i >= n) return;
    const int x = i % Xo, y = (i / Xo) % Yo, z = i / (Xo * Yo);
    float m = 0.f;
    for (int dz = 0; dz < 2; ++dz)
        for (int dy = 0; dy < 2; ++dy)
            for (int dx = 0; dx < 2; ++dx)
                m = fmaxf(m, s[((z * 2 + dz) * Y + y * 2 + dy) * X + x * 2 + dx]);
    d[i] = m;
}

__global__ __launch_bounds__(256) void k_count(const float* m, float* cnt, int n) {
    __shared__ float s[256];
    float a = 0.f;
    for (int i = blockIdx.x * 256 + threadIdx.x; i < n; i += gridDim.x * 256)
        a += m[i];
    s[threadIdx.x] = a;
    __syncthreads();
    for (int t = 128; t > 0; t >>= 1) {
        if (threadIdx.x < t) s[threadIdx.x] += s[threadIdx.x + t];
        __syncthreads();
    }
    if (threadIdx.x == 0) atomicAdd(cnt, s[0]);
}

__global__ void k_to_bf16(const float* src, __bf16* dst, size_t n,
                          const float* mf, int cin) {
    size_t i = (size_t)blockIdx.x * blockDim.x + threadIdx.x;
    if (i >= n) return;
    float v = src[i];
    if (mf) v *= mf[i / (size_t)cin];
    dst[i] = (__bf16)v;
}

// block = one voxel row, thread = channel
__global__ void k_copy_ch(const float* src, float* dst, int cs, int cw,
                          int off) {
    const int v = blockIdx.x;
    const int c = threadIdx.x;
    if (c < cs) dst[(size_t)v * cw + off + c] = src[(size_t)v * cs + c];
}

// masked BN partial sums: coalesced row reads, LDS float atomics, then one
// global atomic per channel per block.
__global__ __launch_bounds__(256) void k_bn_part(const float* y,
                                                 const float* mf, float* acc,
                                                 float* acc2, int nvox,
                                                 int cout) {
    __shared__ float sA[256], sB[256];
    sA[threadIdx.x] = 0.f;
    sB[threadIdx.x] = 0.f;
    __syncthreads();
    for (int v = blockIdx.x; v < nvox; v += gridDim.x) {
        const float  m  = mf[v];
        const float* yr = y + (size_t)v * cout;
        for (int c = threadIdx.x; c < cout; c += 256) {
            const float t = yr[c];
            atomicAdd(&sA[c], m * t);
            atomicAdd(&sB[c], m * t * t);
        }
    }
    __syncthreads();
    if (threadIdx.x < cout) {
        atomicAdd(&acc[threadIdx.x], sA[threadIdx.x]);
        atomicAdd(&acc2[threadIdx.x], sB[threadIdx.x]);
    }
}

__global__ void k_bn_final(const float* acc, const float* acc2,
                           const float* cnt, float* mean, float* var,
                           int cout) {
    const int c = threadIdx.x;
    if (c >= cout) return;
    const float n  = cnt[0];
    const float mu = acc[c] / n;
    mean[c] = mu;
    var[c]  = fmaxf(acc2[c] / n - mu * mu, 0.f);
}

// fused BN (+residual)(+ReLU)(+mask); block = voxel row, thread = channel
__global__ void k_bn_apply(const float* y, const float* mean, const float* var,
                           const float* g, const float* bta, const float* res,
                           const float* mf, float* out, int cout, int relu,
                           int ostride, int ooff) {
    const int v = blockIdx.x;
    const int c = threadIdx.x;
    if (c >= cout) return;
    const size_t i = (size_t)v * cout + c;
    float t = (y[i] - mean[c]) * rsqrtf(var[c] + 1e-5f) * g[c] + bta[c];
    if (res) t += res[i];
    if (relu) t = fmaxf(t, 0.f);
    if (mf) t *= mf[v];
    out[(size_t)v * ostride + ooff + c] = t;
}

// ---------------------------------------------------------------------------
// Host orchestration
// ---------------------------------------------------------------------------
extern "C" void kernel_launch(void* const* d_in, const int* in_sizes, int n_in,
                              void* d_out, int out_size, void* d_ws,
                              size_t ws_size, hipStream_t stream) {
    (void)in_sizes; (void)out_size; (void)ws_size;
    if (n_in < 3 || !d_ws) return;

    const float* feats = (const float*)d_in[0];
    const int*   maski = (const int*)d_in[1];

    // ---- parameter leaves in jax pytree order (dict keys sorted) ----
    int idx = 2;
    auto nf = [&]() -> const float* {
        return (idx < n_in) ? (const float*)d_in[idx++] : nullptr;
    };
    struct ResP { const float *b1,*b2,*bs,*g1,*g2,*gs,*w1,*w2,*ws; };
    auto rdres = [&](bool hs) {
        ResP r{};
        r.b1 = nf(); r.b2 = nf();
        if (hs) r.bs = nf();
        r.g1 = nf(); r.g2 = nf();
        if (hs) r.gs = nf();
        r.w1 = nf(); r.w2 = nf();
        r.ws = hs ? nf() : nullptr;
        return r;
    };
    struct EncP { const float *db,*dg,*dw; ResP r1, r2; };
    auto rdenc = [&](bool ws1) {
        EncP e; e.db = nf(); e.dg = nf(); e.dw = nf();
        e.r1 = rdres(ws1); e.r2 = rdres(false);
        return e;
    };
    struct DecP { ResP r1, r2; const float *ub,*ug,*uw; };
    auto rddec = [&]() {
        DecP d; d.r1 = rdres(true); d.r2 = rdres(false);
        d.ub = nf(); d.ug = nf(); d.uw = nf();
        return d;
    };
    EncP s1 = rdenc(false), s2 = rdenc(true), s3 = rdenc(true), s4 = rdenc(true);
    struct StemP { const float *b1,*b2,*g1,*g2,*w1,*w2; } st;
    st.b1 = nf(); st.b2 = nf(); st.g1 = nf(); st.g2 = nf();
    st.w1 = nf(); st.w2 = nf();
    DecP u1 = rddec(), u2 = rddec(), u3 = rddec(), u4 = rddec();

    // ---- level geometry ----
    static const int ZD[5] = {64, 32, 16, 8, 4};
    static const int YD[5] = {64, 32, 16, 8, 4};
    static const int XD[5] = {32, 16, 8, 4, 2};
    static const int NV[5] = {131072, 16384, 2048, 256, 32};
    const int MAXV = 131072;

    // ---- workspace bump allocator ----
    char*  ws  = (char*)d_ws;
    size_t off = 0;
    auto alloc = [&](size_t bytes) -> void* {
        void* p = ws + off;
        off += (bytes + 255) & ~(size_t)255;
        return p;
    };
    float* MF[5];
    for (int l = 0; l < 5; ++l) MF[l] = (float*)alloc((size_t)NV[l] * 4);
    float*  CNT  = (float*)alloc(64);
    float*  MEAN = (float*)alloc(1024);
    float*  VAR  = (float*)alloc(1024);
    float*  ACC  = (float*)alloc(1024);
    float*  ACC2 = (float*)alloc(1024);
    __bf16* WPK  = (__bf16*)alloc((size_t)8 << 20);
    __bf16* XB   = (__bf16*)alloc((size_t)MAXV * 128 * 2);
    __bf16* AB   = (__bf16*)alloc((size_t)MAXV * 96 * 2);
    float*  T1   = (float*)alloc((size_t)MAXV * 96 * 4);
    float*  T2   = (float*)alloc((size_t)MAXV * 96 * 4);
    float*  A1   = (float*)alloc((size_t)MAXV * 96 * 4);
    float*  WIDE = (float*)alloc((size_t)MAXV * 128 * 4);
    float*  X0 = (float*)alloc((size_t)NV[0] * 32 * 4);
    float*  X1 = (float*)alloc((size_t)NV[1] * 32 * 4);
    float*  X2 = (float*)alloc((size_t)NV[2] * 64 * 4);
    float*  X3 = (float*)alloc((size_t)NV[3] * 128 * 4);
    float*  X4 = (float*)alloc((size_t)NV[4] * 256 * 4);
    float*  Y1 = (float*)alloc((size_t)NV[3] * 256 * 4);
    float*  Y2 = (float*)alloc((size_t)NV[2] * 128 * 4);
    float*  Y3 = (float*)alloc((size_t)NV[1] * 96 * 4);

    // ---- launch helpers ----
    auto to_bf16 = [&](__bf16* dst, const float* src, size_t n, const float* mf,
                       int cin) {
        k_to_bf16<<<dim3((unsigned)((n + 255) / 256)), 256, 0, stream>>>(
            src, dst, n, mf, cin);
    };
    auto pack = [&](const float* w, int ktot, int cout, __bf16* dst) {
        const int    kch   = (ktot + 31) / 32;
        const size_t total = (size_t)kch * (cout / 16) * 512;
        k_pack_w<<<dim3((unsigned)((total + 255) / 256)), 256, 0, stream>>>(
            w, dst, ktot, kch, cout);
    };
    auto fillP = [&](const __bf16* xb, const __bf16* wpk, float* y, int li,
                     int lo, int cin, int cout, int ks, int stride, int pad) {
        ConvP p{};
        p.x = xb; p.wpk = wpk; p.y = y;
        p.Zi = ZD[li]; p.Yi = YD[li]; p.Xi = XD[li];
        p.Zo = ZD[lo]; p.Yo = YD[lo]; p.Xo = XD[lo];
        p.cin = cin; p.cout = cout; p.ks = ks; p.stride = stride; p.pad = pad;
        p.cchunks = cin / 32;
        p.ktot = ks * ks * ks * cin;
        p.kchunks = (p.ktot + 31) / 32;
        p.ntiles = cout / 16;
        return p;
    };
    auto conv = [&](const __bf16* xb, const __bf16* wpk, float* y, int li,
                    int lo, int cin, int cout, int ks, int stride, int pad) {
        ConvP p = fillP(xb, wpk, y, li, lo, cin, cout, ks, stride, pad);
        if ((cin & 31) == 0)
            k_conv_fast<false>
                <<<dim3(NV[lo] / 32, cout / 32), 32, 0, stream>>>(p);
        else
            k_conv_generic<<<dim3(NV[lo] / 16, cout / 32), 32, 0, stream>>>(p);
    };
    auto deconv = [&](const __bf16* xb, float* y, const float* w, int lc,
                      int lf, int cin, int cout) {
        const size_t psz = (size_t)(cin / 32) * (cout / 16) * 512;
        for (int t = 0; t < 8; ++t)
            pack(w + (size_t)t * cin * cout, cin, cout, WPK + (size_t)t * psz);
        for (int t = 0; t < 8; ++t) {
            ConvP p = fillP(xb, WPK + (size_t)t * psz, y, lc, lf, cin, cout, 2,
                            2, 0);
            p.tz = t >> 2; p.ty = (t >> 1) & 1; p.tx = t & 1;
            k_conv_fast<true>
                <<<dim3(NV[lc] / 32, cout / 32), 32, 0, stream>>>(p);
        }
    };
    auto stats = [&](const float* y, int lvl, int cout) {
        hipMemsetAsync(ACC, 0, (size_t)cout * 4, stream);
        hipMemsetAsync(ACC2, 0, (size_t)cout * 4, stream);
        const int blocks = NV[lvl] < 1024 ? NV[lvl] : 1024;
        k_bn_part<<<dim3(blocks), 256, 0, stream>>>(y, MF[lvl], ACC, ACC2,
                                                    NV[lvl], cout);
        k_bn_final<<<dim3(1), 256, 0, stream>>>(ACC, ACC2, CNT + lvl, MEAN, VAR,
                                                cout);
    };
    auto apply = [&](const float* y, const float* g, const float* b,
                     const float* res, const float* mf, float* out, int lvl,
                     int cout, int relu, int ostride, int ooff) {
        k_bn_apply<<<dim3(NV[lvl]), cout, 0, stream>>>(
            y, MEAN, VAR, g, b, res, mf, out, cout, relu, ostride, ooff);
    };

    // conv + BN + ReLU + mask (maskmul multiplies the INPUT by mask)
    auto cbr = [&](const float* x, int cin, const float* w, const float* g,
                   const float* b, int li, int lo, int ks, int stride, int pad,
                   int cout, float* out, const float* maskmul) {
        to_bf16(XB, x, (size_t)NV[li] * cin, maskmul, cin);
        pack(w, ks * ks * ks * cin, cout, WPK);
        conv(XB, WPK, T1, li, lo, cin, cout, ks, stride, pad);
        stats(T1, lo, cout);
        apply(T1, g, b, nullptr, MF[lo], out, lo, cout, 1, cout, 0);
    };

    // residual block (in-place out==x is safe: epilogue is elementwise)
    auto resblk = [&](const float* x, int cin, int cout, int lvl, const ResP& r,
                      float* out) {
        to_bf16(XB, x, (size_t)NV[lvl] * cin, nullptr, cin);
        pack(r.w1, 27 * cin, cout, WPK);
        conv(XB, WPK, T1, lvl, lvl, cin, cout, 3, 1, 1);
        stats(T1, lvl, cout);
        apply(T1, r.g1, r.b1, nullptr, MF[lvl], A1, lvl, cout, 1, cout, 0);
        to_bf16(AB, A1, (size_t)NV[lvl] * cout, nullptr, cout);
        pack(r.w2, 27 * cout, cout, WPK);
        conv(AB, WPK, T2, lvl, lvl, cout, cout, 3, 1, 1);
        const float* sc = x;
        if (r.ws) {  // 1x1x1 projection shortcut + BN
            pack(r.ws, cin, cout, WPK);
            conv(XB, WPK, T1, lvl, lvl, cin, cout, 1, 1, 0);
            stats(T1, lvl, cout);
            apply(T1, r.gs, r.bs, nullptr, nullptr, T1, lvl, cout, 0, cout, 0);
            sc = T1;
        }
        stats(T2, lvl, cout);
        apply(T2, r.g2, r.b2, sc, MF[lvl], out, lvl, cout, 1, cout, 0);
    };

    auto encblk = [&](const float* x, int cin, int cout, int li, int lo,
                      const EncP& e, float* out) {
        cbr(x, cin, e.dw, e.dg, e.db, li, lo, 2, 2, 0, cin, WIDE, nullptr);
        resblk(WIDE, cin, cout, lo, e.r1, out);
        resblk(out, cout, cout, lo, e.r2, out);
    };

    auto decblk = [&](const float* x, int cin, const float* skip, int cs,
                      int cout, int lc, int lf, const DecP& d, float* out) {
        to_bf16(XB, x, (size_t)NV[lc] * cin, nullptr, cin);
        deconv(XB, T1, d.uw, lc, lf, cin, cout);
        stats(T1, lf, cout);
        const int cw = cout + cs;
        apply(T1, d.ug, d.ub, nullptr, MF[lf], WIDE, lf, cout, 1, cw, 0);
        k_copy_ch<<<dim3(NV[lf]), cs, 0, stream>>>(skip, WIDE, cs, cw, cout);
        resblk(WIDE, cw, cout, lf, d.r1, out);
        resblk(out, cout, cout, lf, d.r2, out);
    };

    // ---- mask hierarchy + per-level occupied-voxel counts ----
    k_mask_to_float<<<dim3((NV[0] + 255) / 256), 256, 0, stream>>>(maski, MF[0],
                                                                   NV[0]);
    for (int l = 0; l < 4; ++l) {
        k_pool2<<<dim3((NV[l + 1] + 255) / 256), 256, 0, stream>>>(
            MF[l], MF[l + 1], ZD[l], YD[l], XD[l]);
    }
    hipMemsetAsync(CNT, 0, 64, stream);
    for (int l = 0; l < 5; ++l) {
        int blocks = (NV[l] + 255) / 256;
        if (blocks > 64) blocks = 64;
        k_count<<<dim3(blocks), 256, 0, stream>>>(MF[l], CNT + l, NV[l]);
    }

    // ---- network ----
    cbr(feats, 4, st.w1, st.g1, st.b1, 0, 0, 3, 1, 1, 32, WIDE, MF[0]);
    cbr(WIDE, 32, st.w2, st.g2, st.b2, 0, 0, 3, 1, 1, 32, X0, nullptr);

    encblk(X0, 32, 32, 0, 1, s1, X1);
    encblk(X1, 32, 64, 1, 2, s2, X2);
    encblk(X2, 64, 128, 2, 3, s3, X3);
    encblk(X3, 128, 256, 3, 4, s4, X4);

    decblk(X4, 256, X3, 128, 256, 4, 3, u1, Y1);
    decblk(Y1, 256, X2, 64, 128, 3, 2, u2, Y2);
    decblk(Y2, 128, X1, 32, 96, 2, 1, u3, Y3);
    decblk(Y3, 96, X0, 32, 96, 1, 0, u4, (float*)d_out);
}
